// ISTFT_53068615909609
// MI455X (gfx1250) — matrix operand order repaired and checked by reading, not compiled, not run on hardware
//
#include <hip/hip_runtime.h>
#include <hip/hip_bf16.h>

#define NFFT    1024
#define HOP     256
#define BS      8
#define NFRAMES 2000
#define NFREQ   513      // NFFT/2+1
#define KREAL   513
#define KTOT    1026     // 513 real + 513 imag coefficients
#define KPAD    1056     // 33 * 32  (K tiles of 32 for f16 WMMA)
#define NKT     (KPAD / 32)
#define TPAD    2048     // frames padded for aligned tiles
#define OUTSAMP 512768   // (NFRAMES-1)*HOP + NFFT
#define OUTLEN  511744   // OUTSAMP - NFFT   (trim NFFT/2 each side)

typedef __attribute__((ext_vector_type(16))) _Float16 v16h;
typedef __attribute__((ext_vector_type(8)))  _Float16 v8h;
typedef __attribute__((ext_vector_type(8)))  float    v8f;
typedef __attribute__((ext_vector_type(4)))  unsigned int v4u;

// ---------------------------------------------------------------------------
// Pack folded (Hermitian-symmetry absorbed) inverse-DFT weights, f32 -> f16.
// Aw layout: [o][k], o in [0,1024), k in [0,KPAD). k<513: real coeff for bin k;
// 513<=k<1026: imag coeff for bin k-513; rest zero-pad.
// ---------------------------------------------------------------------------
__global__ void pack_weights(const float* __restrict__ Wr, const float* __restrict__ Wi,
                             _Float16* __restrict__ Aw) {
    int idx = blockIdx.x * blockDim.x + threadIdx.x;
    if (idx >= 1024 * KPAD) return;
    int o = idx / KPAD;
    int k = idx - o * KPAD;
    float v = 0.0f;
    if (k == 0)            v = Wr[o];
    else if (k < 512)      v = Wr[k * 1024 + o] + Wr[(1024 - k) * 1024 + o];
    else if (k == 512)     v = Wr[512 * 1024 + o];
    else if (k == 513)     v = -Wi[o];
    else if (k < 1025)   { int i = k - 513; v = Wi[(1024 - i) * 1024 + o] - Wi[i * 1024 + o]; }
    else if (k == 1025)    v = -Wi[512 * 1024 + o];
    Aw[idx] = (_Float16)v;
}

// ---------------------------------------------------------------------------
// Pack spectra into X[b][k][t], f16, t padded to TPAD (zeros), k padded to KPAD.
// ---------------------------------------------------------------------------
__global__ void pack_spectra(const float* __restrict__ re, const float* __restrict__ im,
                             _Float16* __restrict__ X) {
    size_t idx = (size_t)blockIdx.x * blockDim.x + threadIdx.x;
    if (idx >= (size_t)BS * KPAD * TPAD) return;
    int    t = (int)(idx % TPAD);
    size_t r = idx / TPAD;
    int    k = (int)(r % KPAD);
    int    b = (int)(r / KPAD);
    float v = 0.0f;
    if (t < NFRAMES) {
        if (k < KREAL)     v = re[((size_t)b * NFREQ + k) * NFRAMES + t];
        else if (k < KTOT) v = im[((size_t)b * NFREQ + (k - KREAL)) * NFRAMES + t];
    }
    X[idx] = (_Float16)v;
}

// Concatenate two half8 into a half16 fragment.
static __device__ inline v16h comb(v8h lo, v8h hi) {
    return __builtin_shufflevector(lo, hi, 0, 1, 2, 3, 4, 5, 6, 7,
                                           8, 9, 10, 11, 12, 13, 14, 15);
}

// ---------------------------------------------------------------------------
// GEMM: S[b][t][o] = sum_k Aw[o][k] * X[b][k][t]   (f32 accumulate via WMMA)
// Block: 256 threads = 8 waves; computes 128(M=o) x 128(N=t) tile.
// Wave (wm,wn): wm in 0..1, wn in 0..3 -> 64x32 sub-tile = 4x2 wmma tiles.
// Double-buffered LDS; global loads for slice kt+1 issued before the WMMA
// burst for slice kt, single barrier per iteration.
// ---------------------------------------------------------------------------
__global__ void __launch_bounds__(256) istft_gemm(const _Float16* __restrict__ Aw,
                                                  const _Float16* __restrict__ X,
                                                  float* __restrict__ S) {
    // A: [m][k], 128 x 32 halfs, row stride 40 halfs (80B, 16B aligned)
    __shared__ __align__(16) _Float16 As[2][128 * 40];
    // B: [k][n], 32 x 128 halfs, row stride 144 halfs (288B, 16B aligned)
    __shared__ __align__(16) _Float16 Bs[2][32 * 144];

    const int tid   = threadIdx.x;
    const int mBase = blockIdx.x * 128;
    const int tBase = blockIdx.y * 128;
    const int bb    = blockIdx.z;
    const int lane  = tid & 31;
    const int wid   = tid >> 5;
    const int wm    = wid & 1;       // 0..1  (M direction)
    const int wn    = wid >> 1;      // 0..3  (N direction)
    const int laneh = lane >> 4;     // half-wave id
    const int lm    = lane & 15;

    // Per-thread copy coordinates: 512 16B chunks per operand block.
    int caR[2], caQ[2], cbR[2], cbQ[2];
    const _Float16* aSrc[2];
    const _Float16* bSrc[2];
    #pragma unroll
    for (int cc = 0; cc < 2; ++cc) {
        int c = tid + cc * 256;
        caR[cc] = c >> 2;  caQ[cc] = c & 3;       // A: 128 rows x 4 chunks
        cbR[cc] = c >> 4;  cbQ[cc] = c & 15;      // B: 32 rows  x 16 chunks
        aSrc[cc] = Aw + (size_t)(mBase + caR[cc]) * KPAD + caQ[cc] * 8;
        bSrc[cc] = X + ((size_t)(bb * KPAD + cbR[cc])) * TPAD + tBase + cbQ[cc] * 8;
    }

    v8f acc[4][2];
    #pragma unroll
    for (int i = 0; i < 4; ++i)
        #pragma unroll
        for (int j = 0; j < 2; ++j)
            acc[i][j] = (v8f){0.f, 0.f, 0.f, 0.f, 0.f, 0.f, 0.f, 0.f};

    // ---- prologue: fill buffer 0 with K-slice 0 ----
    v4u stA[2], stB[2];
    #pragma unroll
    for (int cc = 0; cc < 2; ++cc) stA[cc] = *(const v4u*)(aSrc[cc]);
    #pragma unroll
    for (int cc = 0; cc < 2; ++cc) stB[cc] = *(const v4u*)(bSrc[cc]);
    #pragma unroll
    for (int cc = 0; cc < 2; ++cc)
        *(v4u*)(&As[0][caR[cc] * 40 + caQ[cc] * 8]) = stA[cc];
    #pragma unroll
    for (int cc = 0; cc < 2; ++cc)
        *(v4u*)(&Bs[0][cbR[cc] * 144 + cbQ[cc] * 8]) = stB[cc];
    __syncthreads();

    for (int kt = 0; kt < NKT; ++kt) {
        const int p   = kt & 1;
        const int ktn = (kt + 1 < NKT) ? (kt + 1) : kt;  // clamped (uniform EXEC)

        // ---- issue next K-slice global loads (latency overlapped by WMMAs) ----
        #pragma unroll
        for (int cc = 0; cc < 2; ++cc)
            stA[cc] = *(const v4u*)(aSrc[cc] + (size_t)ktn * 32);
        #pragma unroll
        for (int cc = 0; cc < 2; ++cc)
            stB[cc] = *(const v4u*)(bSrc[cc] + (size_t)ktn * 32 * TPAD);

        // ---- fragment gathers from buffer p (ds_load_b128) ----
        const _Float16* Ap = &As[p][0];
        const _Float16* Bp = &Bs[p][0];
        // A 16x32 f16 layout: lane l (M=l&15), VGPR v: K = (v>>2)*16 + (l>>4)*8 + (v&3)*2
        v16h afrag[4];
        #pragma unroll
        for (int i = 0; i < 4; ++i) {
            int m = wm * 64 + i * 16 + lm;
            v8h lo = *(const v8h*)(Ap + m * 40 + laneh * 8);       // K = laneh*8 .. +7
            v8h hi = *(const v8h*)(Ap + m * 40 + 16 + laneh * 8);  // K = 16+laneh*8 .. +7
            afrag[i] = comb(lo, hi);
        }
        // B 32x16 f16 layout: lane l holds column N=l&15; element s: K = laneh*8+s (s<8), 16+laneh*8+(s-8) (s>=8)
        v16h bfrag[2];
        #pragma unroll
        for (int j = 0; j < 2; ++j) {
            int n0 = wn * 32 + j * 16;
            #pragma unroll
            for (int s = 0; s < 8; ++s) {
                bfrag[j][s]     = Bp[(laneh * 8 + s) * 144 + n0 + lm];
                bfrag[j][s + 8] = Bp[(16 + laneh * 8 + s) * 144 + n0 + lm];
            }
        }

        // ---- 8 WMMAs ----
        #pragma unroll
        for (int i = 0; i < 4; ++i)
            #pragma unroll
            for (int j = 0; j < 2; ++j)
                acc[i][j] = __builtin_amdgcn_wmma_f32_16x16x32_f16(
                    false, afrag[i], false, bfrag[j],
                    (short)0, acc[i][j], false, false);

        // ---- stage next slice into the other buffer, then one barrier ----
        _Float16* An = &As[1 - p][0];
        _Float16* Bn = &Bs[1 - p][0];
        #pragma unroll
        for (int cc = 0; cc < 2; ++cc)
            *(v4u*)(An + caR[cc] * 40 + caQ[cc] * 8) = stA[cc];
        #pragma unroll
        for (int cc = 0; cc < 2; ++cc)
            *(v4u*)(Bn + cbR[cc] * 144 + cbQ[cc] * 8) = stB[cc];
        __syncthreads();
    }

    // ---- epilogue: C/D layout: VGPR r, lane l -> M = r + 8*(l>>4), N = l&15 ----
    #pragma unroll
    for (int i = 0; i < 4; ++i) {
        #pragma unroll
        for (int j = 0; j < 2; ++j) {
            #pragma unroll
            for (int r = 0; r < 8; ++r) {
                int o = mBase + wm * 64 + i * 16 + r + 8 * laneh;
                int t = tBase + wn * 32 + j * 16 + lm;
                S[((size_t)bb * TPAD + t) * NFFT + o] = acc[i][j][r];
            }
        }
    }
}

// ---------------------------------------------------------------------------
// Overlap-add + window-sum normalize + trim. Pure gather, no atomics.
// out[b, m] for m in [0, OUTLEN); sample p = m + NFFT/2 gathers <=4 frames.
// ---------------------------------------------------------------------------
__global__ void ola_kernel(const float* __restrict__ S, const float* __restrict__ ola,
                           float* __restrict__ out) {
    size_t idx = (size_t)blockIdx.x * blockDim.x + threadIdx.x;
    if (idx >= (size_t)BS * OUTLEN) return;
    int m = (int)(idx % OUTLEN);
    int b = (int)(idx / OUTLEN);
    int p = m + NFFT / 2;
    int tmn = p - (NFFT - 1);
    tmn = (tmn < 0) ? 0 : ((tmn + HOP - 1) >> 8);
    int tmx = p >> 8;
    if (tmx > NFRAMES - 1) tmx = NFRAMES - 1;
    float acc = 0.0f, ws = 0.0f;
    for (int t = tmn; t <= tmx; ++t) {
        int n = p - (t << 8);
        acc += S[((size_t)b * TPAD + t) * NFFT + n];
        ws  += ola[n];
    }
    ws = (ws < 1e-11f) ? 1e-11f : ws;
    out[idx] = acc / ws;
}

// ---------------------------------------------------------------------------
extern "C" void kernel_launch(void* const* d_in, const int* in_sizes, int n_in,
                              void* d_out, int out_size, void* d_ws, size_t ws_size,
                              hipStream_t stream) {
    (void)in_sizes; (void)n_in; (void)out_size; (void)ws_size;
    const float* re  = (const float*)d_in[0];   // (8, 513, 2000)
    const float* im  = (const float*)d_in[1];   // (8, 513, 2000)
    const float* Wr  = (const float*)d_in[2];   // (1024, 1024) [i][o]
    const float* Wi  = (const float*)d_in[3];   // (1024, 1024) [i][o]
    const float* ola = (const float*)d_in[4];   // (1024,)
    float* out = (float*)d_out;                 // (8, 511744)

    char* ws = (char*)d_ws;
    const size_t AW_BYTES = (size_t)1024 * KPAD * sizeof(_Float16);        //  ~2.2 MB
    const size_t X_BYTES  = (size_t)BS * KPAD * TPAD * sizeof(_Float16);   // ~34.6 MB
    _Float16* Aw = (_Float16*)ws;
    _Float16* X  = (_Float16*)(ws + AW_BYTES);
    float*    S  = (float*)(ws + AW_BYTES + X_BYTES);                      // ~67.1 MB

    {
        int n = 1024 * KPAD;
        pack_weights<<<(n + 255) / 256, 256, 0, stream>>>(Wr, Wi, Aw);
    }
    {
        size_t n = (size_t)BS * KPAD * TPAD;
        pack_spectra<<<(unsigned)((n + 255) / 256), 256, 0, stream>>>(re, im, X);
    }
    istft_gemm<<<dim3(1024 / 128, TPAD / 128, BS), 256, 0, stream>>>(Aw, X, S);
    {
        size_t n = (size_t)BS * OUTLEN;
        ola_kernel<<<(unsigned)((n + 255) / 256), 256, 0, stream>>>(S, ola, out);
    }
}
